// StructureAwareAttention_51737176047745
// MI455X (gfx1250) — matrix operand order repaired and checked by reading, not compiled, not run on hardware
//
#include <hip/hip_runtime.h>

// ---------------------------------------------------------------------------
// StructureAwareAttention for MI455X (gfx1250, wave32, WMMA bf16->f32, TDM)
// N=1024 nodes, NE=512, EE=64, H=8, HID=512, D=64
// ---------------------------------------------------------------------------

#define NN   1024
#define NE   512
#define EE   64
#define HH   8
#define HID  512
#define DD   64

typedef __bf16 bf16;
typedef __attribute__((ext_vector_type(16))) __bf16 bf16x16;
typedef __attribute__((ext_vector_type(8)))  float  floatx8;
typedef __attribute__((ext_vector_type(4)))  unsigned u32x4;
typedef __attribute__((ext_vector_type(8)))  int      i32x8;
typedef __attribute__((ext_vector_type(4)))  int      i32x4;

#define WMMA_BF16(a, b, c) \
  __builtin_amdgcn_wmma_f32_16x16x32_bf16(false, (a), false, (b), (short)0, (c), false, false)

// native bf16 converts (gfx1250 has hw cvt; lets pairs pack into v_cvt_pk_bf16_f32)
__device__ __forceinline__ bf16  f2bf(float f) { return (bf16)f; }
__device__ __forceinline__ float bf2f(bf16 v)  { return (float)v; }

// LDS byte offset of a __shared__ object (generic -> addrspace(3) -> 32-bit offset)
__device__ __forceinline__ unsigned lds_addr_of(const void* p) {
  return (unsigned)(unsigned long long)(__attribute__((address_space(3))) const void*)p;
}

// CDNA5 16-bit A-matrix 16x32 fragment: lane m=L&15; K pair base per VGPR v
__device__ __forceinline__ int a_kidx(int lane, int v) {
  int khalf = (lane >> 4) & 1;                     // lanes 16-31 hold K half +8
  return ((v < 4) ? 0 : 16) + khalf * 8 + (v & 3) * 2;
}
// CDNA5 16-bit B-matrix 32x16 fragment: lane n=L&15; same K order as A (lanes 16-31 hold K half +8)
__device__ __forceinline__ int b_kidx(int lane, int v) {
  int ksec = (lane >> 4) & 1;
  return ((v < 4) ? 0 : 16) + ksec * 8 + (v & 3) * 2;
}

// ---------------------------------------------------------------------------
// TDM: DMA one edge tile  edges[i0 .. i0+15][j0 .. j0+31][0..63]  (fp32)
// into LDS.  2D descriptor: tile_dim0 = 32*64 = 2048 contiguous elements,
// tile_dim1 = 16 rows, row stride = N*EE = 65536 elements, data_size = 4B.
// This toolchain exposes the 6-arg builtin: (g0, g1, g2, g3, g_extra, cpol).
// ---------------------------------------------------------------------------
__device__ __forceinline__ void tdm_load_tile(const float* gsrc, unsigned lds_byte_off) {
  unsigned long long ga = (unsigned long long)(const void*)gsrc;
  u32x4 g0;
  g0[0] = 1u;                                            // count=1 (valid), user mode
  g0[1] = lds_byte_off;                                  // lds_addr
  g0[2] = (unsigned)(ga & 0xFFFFFFFFu);                  // global_addr[31:0]
  g0[3] = (unsigned)((ga >> 32) & 0x01FFFFFFu) | 0x80000000u; // addr[56:32] | type=2
  i32x8 g1;
  g1[0] = 0x00020000;                                    // wg_mask=0, data_size=2 (4B)
  g1[1] = 0;                                             // no atomic barrier; dim0 lo16=0
  g1[2] = 0x04000001;                                    // tensor_dim0=65536, tensor_dim1=1024
  g1[3] = 0x08000000;                                    // tile_dim0 = 2048
  g1[4] = 0x00000010;                                    // tile_dim1 = 16, tile_dim2 = 0
  g1[5] = 0x00010000;                                    // tensor_dim0_stride = 65536
  g1[6] = 0;                                             // stride hi / dim1_stride
  g1[7] = 0;
  i32x4 z4 = {0, 0, 0, 0};                               // groups 2/3: NULL (2D tensor)
  i32x8 z8 = {0, 0, 0, 0, 0, 0, 0, 0};
  __builtin_amdgcn_tensor_load_to_lds(g0, g1, z4, z4, z8, 0);
}

// ---------------------------------------------------------------------------
// Kernel 1: Q/K/V projections.  C[N,HID] = nodes @ W^T + b   (bf16 WMMA, f32 acc)
// Output layout: [H][N][D] bf16, q pre-scaled by D^-0.5.
// ---------------------------------------------------------------------------
__global__ void __launch_bounds__(256) qkv_kernel(
    const float* __restrict__ nodes,
    const float* __restrict__ Wq, const float* __restrict__ bq,
    const float* __restrict__ Wk, const float* __restrict__ bk,
    const float* __restrict__ Wv, const float* __restrict__ bv,
    bf16* __restrict__ q_ws, bf16* __restrict__ k_ws, bf16* __restrict__ v_ws) {
  const int lane = threadIdx.x & 31;
  const int wave = threadIdx.x >> 5;
  int g = blockIdx.x * 8 + wave;               // 0..6143
  int which = g >> 11;                         // 0=q 1=k 2=v
  int t = g & 2047;
  int rt = t >> 5;                             // row tile 0..63
  int ct = t & 31;                             // col tile 0..31
  const float* W    = (which == 0) ? Wq : ((which == 1) ? Wk : Wv);
  const float* bias = (which == 0) ? bq : ((which == 1) ? bk : bv);
  bf16* outw        = (which == 0) ? q_ws : ((which == 1) ? k_ws : v_ws);
  const float scale = (which == 0) ? 0.125f : 1.0f;   // D^-0.5 = 1/8

  const int mn = lane & 15;
  floatx8 acc = {};
  for (int kc = 0; kc < NE / 32; ++kc) {
    bf16x16 a, b;
    const float* arow = nodes + (size_t)(rt * 16 + mn) * NE + kc * 32;
    const float* brow = W     + (size_t)(ct * 16 + mn) * NE + kc * 32;  // B[k][n] = W[n][k]
#pragma unroll
    for (int v = 0; v < 8; ++v) {
      int ka = a_kidx(lane, v);
      float2 fa = *reinterpret_cast<const float2*>(arow + ka);
      a[2 * v] = f2bf(fa.x); a[2 * v + 1] = f2bf(fa.y);
      int kb = b_kidx(lane, v);
      float2 fb = *reinterpret_cast<const float2*>(brow + kb);
      b[2 * v] = f2bf(fb.x); b[2 * v + 1] = f2bf(fb.y);
    }
    acc = WMMA_BF16(a, b, acc);
  }
  // C layout: VGPR r, lanes0-15 -> M=r; lanes16-31 -> M=r+8; N = lane&15
  int col = ct * 16 + mn;
  float bval = bias[col];
  int h = col >> 6, d = col & 63;
  int rlo = (lane < 16) ? 0 : 8;
#pragma unroll
  for (int r = 0; r < 8; ++r) {
    int row = rt * 16 + rlo + r;
    outw[((size_t)h * NN + row) * DD + d] = f2bf((acc[r] + bval) * scale);
  }
}

// ---------------------------------------------------------------------------
// Kernel 2: fused edge-LN + structure bias + flash attention.
// One block (8 wave32) per 16-row i-tile; stream j in chunks of 32.
// Edge tiles are DMA'd by the Tensor Data Mover into LDS, double-pumped:
// tile jc+1 is in flight while WMMA phases run on tile jc.
// ---------------------------------------------------------------------------
__global__ void __launch_bounds__(256) attn_kernel(
    const float* __restrict__ edge_repr, const int* __restrict__ mask,
    const float* __restrict__ Wsk, const float* __restrict__ Wsv,
    const float* __restrict__ bsv,
    const float* __restrict__ pn_g, const float* __restrict__ pn_b,
    const bf16* __restrict__ q_ws, const bf16* __restrict__ k_ws,
    const bf16* __restrict__ v_ws,
    float* __restrict__ attn_out) {
  // ~272 KB LDS: CDNA5 allows up to 320 KB per workgroup.
  __shared__ float rawL[16][32][EE];      // 128 KB raw fp32 edge tile (TDM dest)
  __shared__ bf16  eL[16][32][EE];        //  64 KB LN'd edge tile (bf16)
  __shared__ bf16  qsL[HH][16][EE];       //  16 KB
  __shared__ bf16  wskL[DD * EE];         //   8 KB
  __shared__ bf16  wsvL[DD * EE];         //   8 KB
  __shared__ float sL[HH][16][32];        //  16 KB raw scores
  __shared__ bf16  pL[HH][16][32];        //   8 KB exp(s-m)
  __shared__ float alphaL[HH][16];
  __shared__ float linvL[HH][16];
  __shared__ bf16  ewL[HH][16][EE];       //  16 KB
  __shared__ float pngL[EE], pnbL[EE];

  const int tid  = threadIdx.x;
  const int lane = tid & 31;
  const int wave = tid >> 5;              // == head for per-head work
  const int i0   = blockIdx.x * 16;
  const unsigned raw_off = lds_addr_of(&rawL[0][0][0]);

  // kick off DMA of the first edge tile (TDM ignores EXEC; issue from one wave)
  if (wave == 0) tdm_load_tile(edge_repr + (size_t)i0 * NN * EE, raw_off);

  // ---- stage small weights (fp32 -> bf16) ----
  for (int idx = tid; idx < DD * EE; idx += 256) {
    wskL[idx] = f2bf(Wsk[idx]);
    wsvL[idx] = f2bf(Wsv[idx]);
  }
  if (tid < EE) { pngL[tid] = pn_g[tid]; pnbL[tid] = pn_b[tid]; }

  // ---- Q fragments for this i-tile (wave = head), K = D = 64 -> 2 chunks ----
  bf16x16 qf[2];
  {
    const bf16* qrow = q_ws + ((size_t)wave * NN + (i0 + (lane & 15))) * DD;
#pragma unroll
    for (int kc = 0; kc < 2; ++kc)
#pragma unroll
      for (int v = 0; v < 8; ++v) {
        int ka = kc * 32 + a_kidx(lane, v);
        qf[kc][2 * v] = qrow[ka]; qf[kc][2 * v + 1] = qrow[ka + 1];
      }
  }
  __syncthreads();

  // ---- qs = q @ Wsk  (per head: M=i16, N=e64, K=d64) ----
#pragma unroll
  for (int en = 0; en < 4; ++en) {
    floatx8 acc = {};
#pragma unroll
    for (int kc = 0; kc < 2; ++kc) {
      bf16x16 b;
      int nnc = en * 16 + (lane & 15);
#pragma unroll
      for (int v = 0; v < 8; ++v) {
        int kb = kc * 32 + b_kidx(lane, v);     // k = d, B[d][e] = Wsk[d*EE+e]
        b[2 * v]     = wskL[kb * EE + nnc];
        b[2 * v + 1] = wskL[(kb + 1) * EE + nnc];
      }
      acc = WMMA_BF16(qf[kc], b, acc);
    }
    int rlo = (lane < 16) ? 0 : 8;
    int e = en * 16 + (lane & 15);
#pragma unroll
    for (int r = 0; r < 8; ++r) qsL[wave][rlo + r][e] = f2bf(acc[r]);
  }

  floatx8 o_acc[4] = {};           // head=wave : [M=i16][N=d64]
  floatx8 ew_acc[2][4] = {};       // i=wave*2+pi : [M=h16(8 valid)][N=e64]
  float m_run = -3.0e38f, l_run = 0.0f;
  const int srow_h = tid >> 4, srow_i = tid & 15;   // softmax row owner (tid<128)

  for (int jc = 0; jc < NN / 32; ++jc) {
    const int j0 = jc * 32;

    // ---- wait for the DMA'd tile, then make it visible to the workgroup ----
    if (wave == 0) __builtin_amdgcn_s_wait_tensorcnt(0);
    __syncthreads();

    // ---- Phase A: LayerNorm rawL -> eL (bf16) ----
#pragma unroll
    for (int pp = 0; pp < 2; ++pp) {
      int pr = tid * 2 + pp;               // 512 (i,j) pairs
      int ei = pr >> 5, ej = pr & 31;
      const float4* src = reinterpret_cast<const float4*>(&rawL[ei][ej][0]);
      float sum = 0.f, sq = 0.f;
#pragma unroll
      for (int q4 = 0; q4 < 16; ++q4) {
        float4 f = src[q4];
        sum += f.x + f.y + f.z + f.w;
        sq  += f.x * f.x + f.y * f.y + f.z * f.z + f.w * f.w;
      }
      float mean = sum * (1.0f / EE);
      float var  = sq  * (1.0f / EE) - mean * mean;
      float rs   = rsqrtf(var + 1e-5f);
#pragma unroll
      for (int e = 0; e < EE; ++e) {
        float x = rawL[ei][ej][e];
        eL[ei][ej][e] = f2bf((x - mean) * rs * pngL[e] + pnbL[e]);
      }
    }
    __syncthreads();

    // ---- prefetch next tile while phases B..E compute on this one ----
    if (jc + 1 < NN / 32 && wave == 0)
      tdm_load_tile(edge_repr + ((size_t)i0 * NN + (size_t)(jc + 1) * 32) * EE, raw_off);

    // ---- Phase B: node-node scores, head = wave  (M=i, N=j32, K=d64)
#pragma unroll
    for (int jn = 0; jn < 2; ++jn) {
      floatx8 acc = {};
#pragma unroll
      for (int kc = 0; kc < 2; ++kc) {
        bf16x16 b;
        const bf16* krow = k_ws + ((size_t)wave * NN + j0 + jn * 16 + (lane & 15)) * DD + kc * 32;
#pragma unroll
        for (int v = 0; v < 8; ++v) {
          int kb = b_kidx(lane, v);          // B[d][j] = k[j][d], pairs contiguous
          b[2 * v] = krow[kb]; b[2 * v + 1] = krow[kb + 1];
        }
        acc = WMMA_BF16(qf[kc], b, acc);
      }
      int rlo = (lane < 16) ? 0 : 8;
      int jj = jn * 16 + (lane & 15);
#pragma unroll
      for (int r = 0; r < 8; ++r) sL[wave][rlo + r][jj] = acc[r];
    }
    __syncthreads();

    // ---- Phase C: structure bias, wave owns i = wave*2+pi  (M=h8, N=j32, K=e64)
#pragma unroll
    for (int pi = 0; pi < 2; ++pi) {
      int ii = wave * 2 + pi;
#pragma unroll
      for (int jn = 0; jn < 2; ++jn) {
        floatx8 acc = {};
#pragma unroll
        for (int kc = 0; kc < 2; ++kc) {
          bf16x16 a, b;
          int mh = lane & 7;                 // rows 8..15 duplicated (ignored)
          int nnc = jn * 16 + (lane & 15);
#pragma unroll
          for (int v = 0; v < 8; ++v) {
            int ka = kc * 32 + a_kidx(lane, v);
            a[2 * v] = qsL[mh][ii][ka]; a[2 * v + 1] = qsL[mh][ii][ka + 1];
            int kb = kc * 32 + b_kidx(lane, v);
            b[2 * v] = eL[ii][nnc][kb]; b[2 * v + 1] = eL[ii][nnc][kb + 1];
          }
          acc = WMMA_BF16(a, b, acc);
        }
        if (lane < 16) {                     // lanes0-15 carry M=h=0..7 (valid)
          int jj = jn * 16 + lane;
#pragma unroll
          for (int r = 0; r < 8; ++r) sL[r][ii][jj] += acc[r];
        }
      }
    }
    __syncthreads();

    // ---- Phase D: mask + online softmax, thread t<128 owns row (h,i)
    if (tid < 128) {
      const int* mrow = mask + (size_t)(i0 + srow_i) * NN + j0;
      float sv[32];
      float cmax = -3.0e38f;
#pragma unroll
      for (int jj = 0; jj < 32; ++jj) {
        float x = sL[srow_h][srow_i][jj];
        if (mrow[jj] == 0) x = -1.0e9f;
        sv[jj] = x;
        cmax = fmaxf(cmax, x);
      }
      float mnew = fmaxf(m_run, cmax);
      float al = __expf(m_run - mnew);
      float psum = 0.f;
#pragma unroll
      for (int jj = 0; jj < 32; ++jj) {
        float p = __expf(sv[jj] - mnew);
        psum += p;
        pL[srow_h][srow_i][jj] = f2bf(p);
      }
      l_run = l_run * al + psum;
      m_run = mnew;
      alphaL[srow_h][srow_i] = al;
    }
    __syncthreads();

    // ---- Phase E1: rescale + P@V for head=wave  (M=i, N=d64, K=j32)
    {
      int rlo = (lane < 16) ? 0 : 8;
      float av[8];
#pragma unroll
      for (int r = 0; r < 8; ++r) av[r] = alphaL[wave][rlo + r];
#pragma unroll
      for (int dn = 0; dn < 4; ++dn)
#pragma unroll
        for (int r = 0; r < 8; ++r) o_acc[dn][r] *= av[r];

      bf16x16 pa;
#pragma unroll
      for (int v = 0; v < 8; ++v) {
        int ka = a_kidx(lane, v);
        pa[2 * v] = pL[wave][lane & 15][ka]; pa[2 * v + 1] = pL[wave][lane & 15][ka + 1];
      }
#pragma unroll
      for (int dn = 0; dn < 4; ++dn) {
        bf16x16 b;
        int nnc = dn * 16 + (lane & 15);
#pragma unroll
        for (int v = 0; v < 8; ++v) {
          int kb = b_kidx(lane, v);          // B[j][d] = v[j0+j][d]
          b[2 * v]     = v_ws[((size_t)wave * NN + j0 + kb)     * DD + nnc];
          b[2 * v + 1] = v_ws[((size_t)wave * NN + j0 + kb + 1) * DD + nnc];
        }
        o_acc[dn] = WMMA_BF16(pa, b, o_acc[dn]);
      }
    }

    // ---- Phase E2: ew accumulation for i = wave*2+pi  (M=h8, N=e64, K=j32)
#pragma unroll
    for (int pi = 0; pi < 2; ++pi) {
      int ii = wave * 2 + pi;
      int hlo = (lane < 16) ? 0 : 8;
      float av[8];
#pragma unroll
      for (int r = 0; r < 8; ++r) av[r] = alphaL[(hlo + r) & 7][ii];
#pragma unroll
      for (int en = 0; en < 4; ++en)
#pragma unroll
        for (int r = 0; r < 8; ++r) ew_acc[pi][en][r] *= av[r];

      bf16x16 pa;
      int mh = lane & 7;
#pragma unroll
      for (int v = 0; v < 8; ++v) {
        int ka = a_kidx(lane, v);
        pa[2 * v] = pL[mh][ii][ka]; pa[2 * v + 1] = pL[mh][ii][ka + 1];
      }
#pragma unroll
      for (int en = 0; en < 4; ++en) {
        bf16x16 b;
        int nnc = en * 16 + (lane & 15);
#pragma unroll
        for (int v = 0; v < 8; ++v) {
          int kb = b_kidx(lane, v);          // B[j][e] = eL[ii][j][e]
          b[2 * v] = eL[ii][kb][nnc]; b[2 * v + 1] = eL[ii][kb + 1][nnc];
        }
        ew_acc[pi][en] = WMMA_BF16(pa, b, ew_acc[pi][en]);
      }
    }
    __syncthreads();      // protects eL/sL/pL for next iteration
  }

  // ---- Epilogue ----
  if (tid < 128) linvL[srow_h][srow_i] = 1.0f / l_run;
  __syncthreads();

  // stash normalized ew (x 1/l) as bf16
#pragma unroll
  for (int pi = 0; pi < 2; ++pi) {
    int ii = wave * 2 + pi;
    if (lane < 16) {
#pragma unroll
      for (int en = 0; en < 4; ++en) {
        int e = en * 16 + lane;
#pragma unroll
        for (int r = 0; r < 8; ++r)          // M=r == head r (lanes0-15)
          ewL[r][ii][e] = f2bf(ew_acc[pi][en][r] * linvL[r][ii]);
      }
    }
  }
  __syncthreads();

  // outS = ew @ Wsv^T ; combine: out = o/l + outS + bsv
  {
    int rlo = (lane < 16) ? 0 : 8;
    float lv[8];
#pragma unroll
    for (int r = 0; r < 8; ++r) lv[r] = linvL[wave][rlo + r];

    bf16x16 af[2];
#pragma unroll
    for (int kc = 0; kc < 2; ++kc)
#pragma unroll
      for (int v = 0; v < 8; ++v) {
        int ka = kc * 32 + a_kidx(lane, v);
        af[kc][2 * v] = ewL[wave][lane & 15][ka];
        af[kc][2 * v + 1] = ewL[wave][lane & 15][ka + 1];
      }
#pragma unroll
    for (int dn = 0; dn < 4; ++dn) {
      floatx8 acc = {};
#pragma unroll
      for (int kc = 0; kc < 2; ++kc) {
        bf16x16 b;
        int nnc = dn * 16 + (lane & 15);
#pragma unroll
        for (int v = 0; v < 8; ++v) {
          int kb = kc * 32 + b_kidx(lane, v);  // B[e][d] = Wsv[d*EE+e]
          b[2 * v]     = wsvL[nnc * EE + kb];
          b[2 * v + 1] = wsvL[nnc * EE + kb + 1];
        }
        acc = WMMA_BF16(af[kc], b, acc);
      }
      float bsvv = bsv[dn * 16 + (lane & 15)];
#pragma unroll
      for (int r = 0; r < 8; ++r) {
        int irow = i0 + rlo + r;
        float val = o_acc[dn][r] * lv[r] + acc[r] + bsvv;
        attn_out[(size_t)irow * HID + wave * DD + dn * 16 + (lane & 15)] = val;
      }
    }
  }
}

// ---------------------------------------------------------------------------
// Kernel 3: y = relu(attn @ Wo^T + bo)   (bf16 WMMA)
// ---------------------------------------------------------------------------
__global__ void __launch_bounds__(256) oproj_kernel(
    const float* __restrict__ x, const float* __restrict__ Wo,
    const float* __restrict__ bo, float* __restrict__ y) {
  const int lane = threadIdx.x & 31;
  const int wave = threadIdx.x >> 5;
  int g = blockIdx.x * 8 + wave;               // 0..2047
  int rt = g >> 5, ct = g & 31;
  const int mn = lane & 15;
  floatx8 acc = {};
  for (int kc = 0; kc < HID / 32; ++kc) {
    bf16x16 a, b;
    const float* arow = x  + (size_t)(rt * 16 + mn) * HID + kc * 32;
    const float* brow = Wo + (size_t)(ct * 16 + mn) * HID + kc * 32;
#pragma unroll
    for (int v = 0; v < 8; ++v) {
      int ka = a_kidx(lane, v);
      float2 fa = *reinterpret_cast<const float2*>(arow + ka);
      a[2 * v] = f2bf(fa.x); a[2 * v + 1] = f2bf(fa.y);
      int kb = b_kidx(lane, v);
      float2 fb = *reinterpret_cast<const float2*>(brow + kb);
      b[2 * v] = f2bf(fb.x); b[2 * v + 1] = f2bf(fb.y);
    }
    acc = WMMA_BF16(a, b, acc);
  }
  float bval = bo[ct * 16 + mn];
  int rlo = (lane < 16) ? 0 : 8;
#pragma unroll
  for (int r = 0; r < 8; ++r) {
    float v = acc[r] + bval;
    y[(size_t)(rt * 16 + rlo + r) * HID + ct * 16 + mn] = (v > 0.f) ? v : 0.f;
  }
}

// ---------------------------------------------------------------------------
// Kernel 4: out = LayerNorm(nodes + y)   (one wave32 per row, shfl reduction)
// ---------------------------------------------------------------------------
__global__ void __launch_bounds__(256) lnres_kernel(
    const float* __restrict__ nodes, const float* __restrict__ y,
    const float* __restrict__ g, const float* __restrict__ b,
    float* __restrict__ out) {
  const int lane = threadIdx.x & 31;
  const int wave = threadIdx.x >> 5;
  const int row = blockIdx.x * 8 + wave;
  const float* xr = nodes + (size_t)row * HID;
  const float* yr = y + (size_t)row * HID;
  float xv[16], sum = 0.f, sq = 0.f;
#pragma unroll
  for (int t = 0; t < 16; ++t) {
    int e = lane + 32 * t;
    float v = xr[e] + yr[e];
    xv[t] = v; sum += v; sq += v * v;
  }
#pragma unroll
  for (int off = 16; off > 0; off >>= 1) {
    sum += __shfl_xor(sum, off, 32);
    sq  += __shfl_xor(sq,  off, 32);
  }
  float mean = sum * (1.0f / HID);
  float var  = sq  * (1.0f / HID) - mean * mean;
  float rs   = rsqrtf(var + 1e-5f);
#pragma unroll
  for (int t = 0; t < 16; ++t) {
    int e = lane + 32 * t;
    out[(size_t)row * HID + e] = (xv[t] - mean) * rs * g[e] + b[e];
  }
}

// ---------------------------------------------------------------------------
extern "C" void kernel_launch(void* const* d_in, const int* in_sizes, int n_in,
                              void* d_out, int out_size, void* d_ws, size_t ws_size,
                              hipStream_t stream) {
  const float* nodes = (const float*)d_in[0];
  const float* edge_repr = (const float*)d_in[1];
  const int*   mask = (const int*)d_in[2];
  const float* Wq = (const float*)d_in[3];  const float* bq = (const float*)d_in[4];
  const float* Wk = (const float*)d_in[5];  const float* bk = (const float*)d_in[6];
  const float* Wv = (const float*)d_in[7];  const float* bv = (const float*)d_in[8];
  const float* Wsk = (const float*)d_in[9];
  // d_in[10] = bsk: provably softmax-invariant (constant per (h,i) across j) -> unused
  const float* Wsv = (const float*)d_in[11]; const float* bsv = (const float*)d_in[12];
  const float* Wo = (const float*)d_in[13]; const float* bo = (const float*)d_in[14];
  const float* pn_g = (const float*)d_in[15]; const float* pn_b = (const float*)d_in[16];
  const float* ln_g = (const float*)d_in[17]; const float* ln_b = (const float*)d_in[18];

  // workspace: 3 MB bf16 q/k/v + 2 MB y fp32; attn output staged in d_out (2 MB)
  char* ws = (char*)d_ws;
  bf16* q_ws = (bf16*)ws;                       // [H][N][D] bf16, 1 MB
  bf16* k_ws = (bf16*)(ws + (1u << 20));
  bf16* v_ws = (bf16*)(ws + (2u << 20));
  float* y   = (float*)(ws + (3u << 20));       // [N][HID] fp32, 2 MB
  float* attn = (float*)d_out;                  // reused as [N][HID] scratch

  qkv_kernel<<<768, 256, 0, stream>>>(nodes, Wq, bq, Wk, bk, Wv, bv, q_ws, k_ws, v_ws);
  attn_kernel<<<64, 256, 0, stream>>>(edge_repr, mask, Wsk, Wsv, bsv, pn_g, pn_b,
                                      q_ws, k_ws, v_ws, attn);
  oproj_kernel<<<256, 256, 0, stream>>>(attn, Wo, bo, y);
  lnres_kernel<<<128, 256, 0, stream>>>(nodes, y, ln_g, ln_b, (float*)d_out);
}
